// GNNPolicy_37237366456826
// MI455X (gfx1250) — matrix operand order repaired and checked by reading, not compiled, not run on hardware
//
#include <hip/hip_runtime.h>
#include <hip/hip_bf16.h>

// MI455X (gfx1250) 2-layer GAT + MLP policy head.
// GEMMs: v_wmma_f32_16x16x32_f16, f32 accumulate. Weights pre-packed into
// B-fragment order (f16, zero-padded) so every fragment load is a vector load.
// Edge softmax-aggregation: wave-per-edge, coalesced gathers + f32 atomics.
// Workspace: 196*N*4 bytes + 64KB fragment area (~78.5 MB for N=100000).

typedef __attribute__((ext_vector_type(16))) _Float16 v16h;
typedef __attribute__((ext_vector_type(8)))  _Float16 v8h;
typedef __attribute__((ext_vector_type(8)))  float    v8f;
typedef __attribute__((ext_vector_type(4)))  float    v4f;

#define HID 64
#define MLPH 128
#define OUTD 74
#define NEG_SLOPE 0.2f

// ---- monotone float <-> uint encoding for atomic float max ----
__device__ __forceinline__ unsigned fenc(float f) {
    unsigned u = __float_as_uint(f);
    return (u & 0x80000000u) ? ~u : (u | 0x80000000u);
}
__device__ __forceinline__ float fdec(unsigned u) {
    u = (u & 0x80000000u) ? (u & 0x7FFFFFFFu) : ~u;
    return __uint_as_float(u);
}

// =====================================================================
// Pack W[Kin x Ncols] (f32) into WMMA B-fragment order (f16, zero padded):
// frag[((kb*CT + ct)*32 + lane)*16 + h] = W[(kb*32 + (h&7) + (lane>>4)*8 + (h>>3)*16)*Ncols
//                                           + ct*16 + (lane&15)]
// =====================================================================
__global__ void pack_wfrag_kernel(const float* __restrict__ W,
                                  _Float16* __restrict__ out,
                                  int Kin, int Ncols, int KB, int CT) {
    int idx = blockIdx.x * blockDim.x + threadIdx.x;
    int total = KB * CT * 512;
    if (idx >= total) return;
    int h    = idx & 15;
    int l    = (idx >> 4) & 31;
    int tile = idx >> 9;
    int ct   = tile % CT;
    int kb   = tile / CT;
    int k    = kb * 32 + (h & 7) + (l >> 4) * 8 + (h >> 3) * 16;
    int col  = ct * 16 + (l & 15);
    out[idx] = (k < Kin && col < Ncols) ? (_Float16)W[k * Ncols + col]
                                        : (_Float16)0.f;
}

// =====================================================================
// GEMM: H[N x 64] = X[N x KIN] @ W[KIN x 64] via pre-packed B fragments.
// One wave per 16-row tile; 4 column tiles of 16.
// =====================================================================
template <int KIN>
__global__ void gat_gemm_xw_kernel(const float* __restrict__ X,
                                   const _Float16* __restrict__ Wfrag,
                                   float* __restrict__ H, int N) {
    constexpr int KB = (KIN + 31) / 32;
    int wave = (blockIdx.x * blockDim.x + threadIdx.x) >> 5;
    int lane = threadIdx.x & 31;
    int base = wave * 16;
    if (base >= N) return;
    int lh = lane >> 4;   // lane half
    int ln = lane & 15;

    v8f C[4] = {};
    int row = base + ln;
    if (row >= N) row = N - 1;

#pragma unroll
    for (int kb = 0; kb < KB; ++kb) {
        // ---- A fragment: two contiguous 8-float runs per 32-K block ----
        v16h a;
        if (KIN >= 32) {
            const v4f* Xr = (const v4f*)(X + (size_t)row * KIN + kb * 32);
            v4f f0 = Xr[lh * 2 + 0];
            v4f f1 = Xr[lh * 2 + 1];
            v4f f2 = Xr[lh * 2 + 4];
            v4f f3 = Xr[lh * 2 + 5];
#pragma unroll
            for (int h = 0; h < 4; ++h) {
                a[h]      = (_Float16)f0[h];
                a[h + 4]  = (_Float16)f1[h];
                a[h + 8]  = (_Float16)f2[h];
                a[h + 12] = (_Float16)f3[h];
            }
        } else {  // KIN = 3: only lanes 0-15 (K=0..7) carry data
            float x0 = X[(size_t)row * KIN + 0];
            float x1 = X[(size_t)row * KIN + 1];
            float x2 = X[(size_t)row * KIN + 2];
            bool lo = (lh == 0);
#pragma unroll
            for (int h = 0; h < 16; ++h) a[h] = (_Float16)0.f;
            a[0] = lo ? (_Float16)x0 : (_Float16)0.f;
            a[1] = lo ? (_Float16)x1 : (_Float16)0.f;
            a[2] = lo ? (_Float16)x2 : (_Float16)0.f;
        }
#pragma unroll
        for (int ct = 0; ct < 4; ++ct) {
            v16h b = *(const v16h*)(Wfrag + ((size_t)(kb * 4 + ct) * 32 + lane) * 16);
            C[ct] = __builtin_amdgcn_wmma_f32_16x16x32_f16(
                false, a, false, b, (short)0, C[ct], false, false);
        }
    }
#pragma unroll
    for (int ct = 0; ct < 4; ++ct) {
#pragma unroll
        for (int r = 0; r < 8; ++r) {
            int orow = base + r + lh * 8;
            if (orow < N) H[(size_t)orow * HID + ct * 16 + ln] = C[ct][r];
        }
    }
}

// =====================================================================
// Per-node attention scores: ssrc[i] = H[i].a_src, sdst[i] = H[i].a_dst
// =====================================================================
__global__ void node_scores_kernel(const float* __restrict__ H,
                                   const float* __restrict__ a_src,
                                   const float* __restrict__ a_dst,
                                   float* __restrict__ ssrc,
                                   float* __restrict__ sdst, int N) {
    int i = blockIdx.x * blockDim.x + threadIdx.x;
    if (i >= N) return;
    const v4f* h  = (const v4f*)(H + (size_t)i * HID);
    const v4f* as = (const v4f*)a_src;
    const v4f* ad = (const v4f*)a_dst;
    float s0 = 0.f, s1 = 0.f;
#pragma unroll
    for (int j = 0; j < HID / 4; ++j) {
        v4f hv = h[j], av = as[j], dv = ad[j];
#pragma unroll
        for (int c = 0; c < 4; ++c) {
            s0 += hv[c] * av[c];
            s1 += hv[c] * dv[c];
        }
    }
    ssrc[i] = s0;
    sdst[i] = s1;
}

// =====================================================================
// Zero accumulator (N*64), reset segment max encoding & sums
// =====================================================================
__global__ void init_layer_kernel(float* __restrict__ Acc,
                                  unsigned* __restrict__ menc,
                                  float* __restrict__ ssum, int N) {
    int i = blockIdx.x * blockDim.x + threadIdx.x;
    if (i < N * HID) Acc[i] = 0.f;
    if (i < N) { menc[i] = 0u; ssum[i] = 0.f; }
}

__device__ __forceinline__ void edge_ids(const int* __restrict__ ei, int E,
                                         int t, int& s, int& d) {
    if (t < E) { s = ei[t]; d = ei[E + t]; }
    else       { s = d = t - E; }   // self-loop
}

__device__ __forceinline__ float leaky(float e) {
    return e >= 0.f ? e : NEG_SLOPE * e;
}

// Pass 1: segment max of leaky-relu edge scores into dst (uint-encoded)
__global__ void edge_max_kernel(const int* __restrict__ ei, int E, int N,
                                const float* __restrict__ ssrc,
                                const float* __restrict__ sdst,
                                unsigned* __restrict__ menc) {
    int t = blockIdx.x * blockDim.x + threadIdx.x;
    if (t >= E + N) return;
    int s, d; edge_ids(ei, E, t, s, d);
    float e = leaky(ssrc[s] + sdst[d]);
    atomicMax(&menc[d], fenc(e));
}

// Pass 2: ssum[d] += exp(e - m[d])
__global__ void edge_expsum_kernel(const int* __restrict__ ei, int E, int N,
                                   const float* __restrict__ ssrc,
                                   const float* __restrict__ sdst,
                                   const unsigned* __restrict__ menc,
                                   float* __restrict__ ssum) {
    int t = blockIdx.x * blockDim.x + threadIdx.x;
    if (t >= E + N) return;
    int s, d; edge_ids(ei, E, t, s, d);
    float e = leaky(ssrc[s] + sdst[d]);
    atomicAdd(&ssum[d], __expf(e - fdec(menc[d])));
}

// Pass 3: Acc[d] += alpha * H[s], one wave per edge (2 floats per lane)
__global__ void edge_agg_kernel(const int* __restrict__ ei, int E, int N,
                                const float* __restrict__ ssrc,
                                const float* __restrict__ sdst,
                                const unsigned* __restrict__ menc,
                                const float* __restrict__ ssum,
                                const float* __restrict__ H,
                                float* __restrict__ Acc) {
    int wave = (blockIdx.x * blockDim.x + threadIdx.x) >> 5;
    int lane = threadIdx.x & 31;
    if (wave >= E + N) return;
    int s, d; edge_ids(ei, E, wave, s, d);
    float e = leaky(ssrc[s] + sdst[d]);
    float alpha = __expf(e - fdec(menc[d])) / ssum[d];
    float2 hv = *(const float2*)(H + (size_t)s * HID + lane * 2);
    atomicAdd(&Acc[(size_t)d * HID + lane * 2 + 0], alpha * hv.x);
    atomicAdd(&Acc[(size_t)d * HID + lane * 2 + 1], alpha * hv.y);
}

// bias + relu, in place
__global__ void bias_relu_kernel(float* __restrict__ Acc,
                                 const float* __restrict__ b, int total) {
    int i = blockIdx.x * blockDim.x + threadIdx.x;
    if (i >= total) return;
    float v = Acc[i] + b[i & (HID - 1)];
    Acc[i] = v > 0.f ? v : 0.f;
}

// =====================================================================
// Fused MLP head: out = relu(X @ Wp1 + bp1) @ Wp2 + bp2
// Wave per 16-node tile; 16x128 hidden staged in LDS (f16) between GEMMs.
// =====================================================================
__global__ void mlp_head_kernel(const float* __restrict__ X,
                                const _Float16* __restrict__ Wp1f,
                                const float* __restrict__ bp1,
                                const _Float16* __restrict__ Wp2f,
                                const float* __restrict__ bp2,
                                float* __restrict__ out, int N) {
    __shared__ _Float16 lds[8][16 * MLPH];   // 4KB/wave, 32KB/WG
    int wid  = threadIdx.x >> 5;
    int lane = threadIdx.x & 31;
    int wave = (blockIdx.x * blockDim.x + threadIdx.x) >> 5;
    int base = wave * 16;
    int lh = lane >> 4, ln = lane & 15;
    bool active = base < N;
    int rowA = active ? base + ln : 0;
    if (rowA >= N) rowA = N - 1;

    // ---- phase 1: hid = relu(X[16x64] @ Wp1[64x128] + bp1) ----
    v8f C1[8] = {};
#pragma unroll
    for (int kb = 0; kb < 2; ++kb) {
        v16h a;
        const v4f* Xr = (const v4f*)(X + (size_t)rowA * HID + kb * 32);
        v4f f0 = Xr[lh * 2 + 0];
        v4f f1 = Xr[lh * 2 + 1];
        v4f f2 = Xr[lh * 2 + 4];
        v4f f3 = Xr[lh * 2 + 5];
#pragma unroll
        for (int h = 0; h < 4; ++h) {
            a[h]      = (_Float16)f0[h];
            a[h + 4]  = (_Float16)f1[h];
            a[h + 8]  = (_Float16)f2[h];
            a[h + 12] = (_Float16)f3[h];
        }
#pragma unroll
        for (int ct = 0; ct < 8; ++ct) {
            v16h b = *(const v16h*)(Wp1f + ((size_t)(kb * 8 + ct) * 32 + lane) * 16);
            C1[ct] = __builtin_amdgcn_wmma_f32_16x16x32_f16(
                false, a, false, b, (short)0, C1[ct], false, false);
        }
    }
#pragma unroll
    for (int ct = 0; ct < 8; ++ct) {
        int col = ct * 16 + ln;
        float bb = bp1[col];
#pragma unroll
        for (int r = 0; r < 8; ++r) {
            float v = C1[ct][r] + bb;
            v = v > 0.f ? v : 0.f;
            lds[wid][(r + lh * 8) * MLPH + col] = (_Float16)v;
        }
    }
    __syncthreads();

    // ---- phase 2: out = hid[16x128] @ Wp2[128x74] + bp2 (cols padded to 80) ----
#pragma unroll
    for (int ct = 0; ct < 5; ++ct) {
        v8f C2 = {};
        int col = ct * 16 + ln;
#pragma unroll
        for (int kb = 0; kb < 4; ++kb) {
            // A fragment from LDS: two contiguous 8-half runs (16B each)
            v8h alo = *(const v8h*)(&lds[wid][ln * MLPH + kb * 32 + lh * 8]);
            v8h ahi = *(const v8h*)(&lds[wid][ln * MLPH + kb * 32 + 16 + lh * 8]);
            v16h a;
#pragma unroll
            for (int h = 0; h < 8; ++h) { a[h] = alo[h]; a[h + 8] = ahi[h]; }
            v16h b = *(const v16h*)(Wp2f + ((size_t)(kb * 5 + ct) * 32 + lane) * 16);
            C2 = __builtin_amdgcn_wmma_f32_16x16x32_f16(
                false, a, false, b, (short)0, C2, false, false);
        }
        if (col < OUTD) {
            float bb = bp2[col];
#pragma unroll
            for (int r = 0; r < 8; ++r) {
                int row = base + r + lh * 8;
                if (active && row < N) out[(size_t)row * OUTD + col] = C2[r] + bb;
            }
        }
    }
}

extern "C" void kernel_launch(void* const* d_in, const int* in_sizes, int n_in,
                              void* d_out, int out_size, void* d_ws, size_t ws_size,
                              hipStream_t stream) {
    const int N = in_sizes[0] / 3;   // IN_DIM = 3
    const int E = in_sizes[1] / 2;
    const int Etot = E + N;

    const float* x   = (const float*)d_in[0];
    const int*   ei  = (const int*)d_in[1];
    const float* W1  = (const float*)d_in[2];
    const float* a1s = (const float*)d_in[3];
    const float* a1d = (const float*)d_in[4];
    const float* b1  = (const float*)d_in[5];
    const float* W2  = (const float*)d_in[6];
    const float* a2s = (const float*)d_in[7];
    const float* a2d = (const float*)d_in[8];
    const float* b2  = (const float*)d_in[9];
    const float* Wp1 = (const float*)d_in[10];
    const float* bp1 = (const float*)d_in[11];
    const float* Wp2 = (const float*)d_in[12];
    const float* bp2 = (const float*)d_in[13];
    float* out = (float*)d_out;

    // workspace layout (floats): H | X2 | X3 | ssrc | sdst | menc | ssum | frags
    float* ws = (float*)d_ws;
    float*    H    = ws;
    float*    X2   = ws + (size_t)N * 64;
    float*    X3   = ws + (size_t)N * 128;
    float*    ssrc = ws + (size_t)N * 192;
    float*    sdst = ws + (size_t)N * 193;
    unsigned* menc = (unsigned*)(ws + (size_t)N * 194);
    float*    ssum = ws + (size_t)N * 195;
    // fragment area, 64B aligned
    size_t fragOff = (((size_t)N * 196 * 4) + 63) & ~(size_t)63;
    _Float16* W1f  = (_Float16*)((char*)d_ws + fragOff);          // 1*4*512
    _Float16* W2f  = W1f + 2048;                                  // 2*4*512
    _Float16* Wp1f = W2f + 4096;                                  // 2*8*512
    _Float16* Wp2f = Wp1f + 8192;                                 // 4*5*512

    const int rowWaves = (N + 15) / 16;
    const int gemmGrid = (rowWaves + 7) / 8;            // 8 waves / block
    const int nodeGrid = (N + 255) / 256;
    const int initGrid = (N * HID + 255) / 256;
    const int edgeGrid = (Etot + 255) / 256;
    const int aggGrid  = (Etot + 7) / 8;                // wave per edge

    // ---- pack all weight matrices into B-fragment order (tiny) ----
    pack_wfrag_kernel<<<(1 * 4 * 512 + 255) / 256, 256, 0, stream>>>(W1,  W1f,  3,   HID,  1, 4);
    pack_wfrag_kernel<<<(2 * 4 * 512 + 255) / 256, 256, 0, stream>>>(W2,  W2f,  HID, HID,  2, 4);
    pack_wfrag_kernel<<<(2 * 8 * 512 + 255) / 256, 256, 0, stream>>>(Wp1, Wp1f, HID, MLPH, 2, 8);
    pack_wfrag_kernel<<<(4 * 5 * 512 + 255) / 256, 256, 0, stream>>>(Wp2, Wp2f, MLPH, OUTD, 4, 5);

    // ---------------- Layer 1 ----------------
    gat_gemm_xw_kernel<3><<<gemmGrid, 256, 0, stream>>>(x, W1f, H, N);
    node_scores_kernel<<<nodeGrid, 256, 0, stream>>>(H, a1s, a1d, ssrc, sdst, N);
    init_layer_kernel<<<initGrid, 256, 0, stream>>>(X2, menc, ssum, N);
    edge_max_kernel<<<edgeGrid, 256, 0, stream>>>(ei, E, N, ssrc, sdst, menc);
    edge_expsum_kernel<<<edgeGrid, 256, 0, stream>>>(ei, E, N, ssrc, sdst, menc, ssum);
    edge_agg_kernel<<<aggGrid, 256, 0, stream>>>(ei, E, N, ssrc, sdst, menc, ssum, H, X2);
    bias_relu_kernel<<<initGrid, 256, 0, stream>>>(X2, b1, N * HID);

    // ---------------- Layer 2 ----------------
    gat_gemm_xw_kernel<HID><<<gemmGrid, 256, 0, stream>>>(X2, W2f, H, N);
    node_scores_kernel<<<nodeGrid, 256, 0, stream>>>(H, a2s, a2d, ssrc, sdst, N);
    init_layer_kernel<<<initGrid, 256, 0, stream>>>(X3, menc, ssum, N);
    edge_max_kernel<<<edgeGrid, 256, 0, stream>>>(ei, E, N, ssrc, sdst, menc);
    edge_expsum_kernel<<<edgeGrid, 256, 0, stream>>>(ei, E, N, ssrc, sdst, menc, ssum);
    edge_agg_kernel<<<aggGrid, 256, 0, stream>>>(ei, E, N, ssrc, sdst, menc, ssum, H, X3);
    bias_relu_kernel<<<initGrid, 256, 0, stream>>>(X3, b2, N * HID);

    // ---------------- MLP head ----------------
    mlp_head_kernel<<<gemmGrid, 256, 0, stream>>>(X3, Wp1f, bp1, Wp2f, bp2, out, N);
}
